// KPConvRegressor_36481452212840
// MI455X (gfx1250) — matrix operand order repaired and checked by reading, not compiled, not run on hardware
//
#include <hip/hip_runtime.h>
#include <hip/hip_bf16.h>
#include <math.h>

#define N_PTS 50000
#define NNB   32
#define KPTS  15
#define KPAD  16
#define D_IN  64
#define D_OUT 1024
#define BATCH 16
#define SIGMA 0.3f
#define TILE  16
#define KK    (KPAD * D_IN)   // 1024 padded inner dim

typedef __attribute__((ext_vector_type(16))) __bf16 v16bf;
typedef __attribute__((ext_vector_type(8)))  float  v8f;
typedef __attribute__((ext_vector_type(8)))  unsigned short v8u;
typedef __attribute__((ext_vector_type(16))) unsigned short v16u;

// round-to-nearest-even f32 -> bf16 (bit level, no __bf16 scalar ops needed)
static __device__ __forceinline__ unsigned short f2bf(float f) {
  unsigned int u = __float_as_uint(f);
  unsigned int r = (u + 0x7FFFu + ((u >> 16) & 1u)) >> 16;
  return (unsigned short)r;
}

// build a v16bf A/B fragment from two 16-byte chunks (LDS: -> 2x ds_load_b128)
static __device__ __forceinline__ v16bf ld_frag2(const unsigned short* p0,
                                                 const unsigned short* p1) {
  v8u lo = *(const v8u*)p0;
  v8u hi = *(const v8u*)p1;
  v16u t;
#pragma unroll
  for (int i = 0; i < 8; ++i) { t[i] = lo[i]; t[i + 8] = hi[i]; }
  return __builtin_bit_cast(v16bf, t);
}

// 32-byte contiguous global fragment (lane = K row of B)
static __device__ __forceinline__ v16bf ld_frag_g(const unsigned short* p) {
  v16u t = *(const v16u*)p;
  return __builtin_bit_cast(v16bf, t);
}

// ---------------- prep kernels ----------------

__global__ void k_feats_bf16(const float* __restrict__ f,
                             unsigned short* __restrict__ o, int n) {
  int i = blockIdx.x * blockDim.x + threadIdx.x;
  if (i < n) o[i] = f2bf(f[i]);
}

// kp_weights [15][64][1024] f32 -> Wb [1024][1024] bf16 (rows 960..1023 zero)
__global__ void k_prep_w(const float* __restrict__ w,
                         unsigned short* __restrict__ o) {
  int i = blockIdx.x * blockDim.x + threadIdx.x;   // KK*D_OUT threads
  int kk = i >> 10;
  float v = (kk < KPTS * D_IN) ? w[i] : 0.0f;      // same flat index while kk<960
  o[i] = f2bf(v);
}

__global__ void k_zero(float* __restrict__ p, int n) {
  int i = blockIdx.x * blockDim.x + threadIdx.x;
  if (i < n) p[i] = 0.0f;
}

// ---------------- fused KPConv + leaky-relu + segment-sum ----------------

__global__ __launch_bounds__(128)
void k_kpconv(const float* __restrict__ pos,
              const float* __restrict__ kp,
              const int*   __restrict__ nidx,
              const int*   __restrict__ batch,
              const unsigned short* __restrict__ featsb,
              const unsigned short* __restrict__ Wb,
              float* __restrict__ pool) {
  __shared__ int   s_idx[TILE][NNB];
  __shared__ float s_ppos[TILE][4];
  __shared__ float s_kp[KPTS][4];
  __shared__ int   s_batch[TILE];
  __shared__ __align__(16) unsigned short s_H[TILE][KPAD][NNB];  // 16 KB
  __shared__ __align__(16) unsigned short s_G[TILE][KK];         // 32 KB

  const int tid  = threadIdx.x;
  const int lane = tid & 31;
  const int wv   = tid >> 5;
  const int p0   = blockIdx.x * TILE;

  if (tid < TILE) {
    s_ppos[tid][0] = pos[(p0 + tid) * 3 + 0];
    s_ppos[tid][1] = pos[(p0 + tid) * 3 + 1];
    s_ppos[tid][2] = pos[(p0 + tid) * 3 + 2];
    s_batch[tid]   = batch[p0 + tid];
  }
  if (tid >= 64 && tid < 64 + KPTS) {
    int k = tid - 64;
    s_kp[k][0] = kp[k * 3 + 0];
    s_kp[k][1] = kp[k * 3 + 1];
    s_kp[k][2] = kp[k * 3 + 2];
  }
  for (int p = tid; p < TILE * NNB; p += 128)
    s_idx[p >> 5][p & 31] = nidx[p0 * NNB + p];
  __syncthreads();

  // ---- phase 1: kernel-point correlation H^T[k][j] per point ----
  for (int p = tid; p < TILE * NNB; p += 128) {
    int m = p >> 5, j = p & 31;
    int gi = s_idx[m][j];
    float rx = pos[gi * 3 + 0] - s_ppos[m][0];
    float ry = pos[gi * 3 + 1] - s_ppos[m][1];
    float rz = pos[gi * 3 + 2] - s_ppos[m][2];
#pragma unroll
    for (int k = 0; k < KPAD; ++k) {
      float h = 0.0f;
      if (k < KPTS) {
        float dx = rx - s_kp[k][0];
        float dy = ry - s_kp[k][1];
        float dz = rz - s_kp[k][2];
        float d = sqrtf(dx * dx + dy * dy + dz * dz);
        h = fmaxf(0.0f, 1.0f - d * (1.0f / SIGMA));
      }
      s_H[m][k][j] = f2bf(h);
    }
  }
  __syncthreads();

  // ---- phase 2: per-point G[16k x 64d] = H^T(16x32) x F(32x64), WMMA ----
  {
    const int kA = lane & 15, hA = lane >> 4;
#pragma unroll
    for (int i = 0; i < 4; ++i) {
      int m = wv * 4 + i;
      v16bf a = ld_frag2(&s_H[m][kA][hA * 8], &s_H[m][kA][16 + hA * 8]);
      const unsigned short* frow = featsb + (size_t)s_idx[m][lane] * D_IN;
#pragma unroll
      for (int dt = 0; dt < 4; ++dt) {
        v16u tb;
#pragma unroll
        for (int s = 0; s < 16; ++s) {
          int j = (s < 8) ? 8 * hA + s : 16 + 8 * hA + (s - 8);
          tb[s] = featsb[(size_t)s_idx[m][j] * D_IN + dt * 16 + kA];
        }
        v16bf b = __builtin_bit_cast(v16bf, tb);
        v8f c = {};
        c = __builtin_amdgcn_wmma_f32_16x16x32_bf16(false, a, false, b,
                                                    (short)0, c, false, false);
#pragma unroll
        for (int r = 0; r < 8; ++r)
          s_G[m][(r + 8 * hA) * D_IN + dt * 16 + kA] = f2bf(c[r]);
      }
    }
  }
  __syncthreads();

  // ---- phase 3: X[16 x 1024] = G[16 x 1024] x Wb[1024 x 1024], WMMA ----
  const int mA = lane & 15, hB = lane >> 4;
  const bool uniform = (s_batch[0] == s_batch[TILE - 1]);  // batch is sorted
  const int  ub = s_batch[0];

#pragma unroll
  for (int g4 = 0; g4 < 4; ++g4) {
    const int nbase = wv * 256 + g4 * 64;
    v8f acc0 = {}, acc1 = {}, acc2 = {}, acc3 = {};
    for (int ks = 0; ks < KK / 32; ++ks) {
      const int K0 = ks * 32;
      v16bf a = ld_frag2(&s_G[mA][K0 + 8 * hB], &s_G[mA][K0 + 16 + 8 * hB]);
      v16u t0, t1, t2, t3;
#pragma unroll
      for (int s = 0; s < 16; ++s) {
        int k = K0 + ((s < 8) ? 8 * hB + s : 16 + 8 * hB + (s - 8));
        const unsigned short* wrow = Wb + (size_t)k * D_OUT + nbase + mA;
        t0[s] = wrow[0]; t1[s] = wrow[16]; t2[s] = wrow[32]; t3[s] = wrow[48];
      }
      v16bf b0 = __builtin_bit_cast(v16bf, t0);
      v16bf b1 = __builtin_bit_cast(v16bf, t1);
      v16bf b2 = __builtin_bit_cast(v16bf, t2);
      v16bf b3 = __builtin_bit_cast(v16bf, t3);
      acc0 = __builtin_amdgcn_wmma_f32_16x16x32_bf16(false, a, false, b0, (short)0, acc0, false, false);
      acc1 = __builtin_amdgcn_wmma_f32_16x16x32_bf16(false, a, false, b1, (short)0, acc1, false, false);
      acc2 = __builtin_amdgcn_wmma_f32_16x16x32_bf16(false, a, false, b2, (short)0, acc2, false, false);
      acc3 = __builtin_amdgcn_wmma_f32_16x16x32_bf16(false, a, false, b3, (short)0, acc3, false, false);
    }
    // fused leaky-relu + global mean-pool accumulation
#pragma unroll
    for (int t = 0; t < 4; ++t) {
      v8f acc = (t == 0) ? acc0 : (t == 1) ? acc1 : (t == 2) ? acc2 : acc3;
      const int n = nbase + t * 16 + mA;
      if (uniform) {
        float s = 0.0f;
#pragma unroll
        for (int r = 0; r < 8; ++r) {
          float x = acc[r];
          s += (x > 0.0f) ? x : 0.1f * x;
        }
        s += __shfl_xor(s, 16, 32);   // combine the two M-halves
        if (lane < 16) atomicAdd(&pool[ub * D_OUT + n], s);
      } else {
#pragma unroll
        for (int r = 0; r < 8; ++r) {
          float x = acc[r];
          x = (x > 0.0f) ? x : 0.1f * x;
          atomicAdd(&pool[s_batch[r + 8 * hB] * D_OUT + n], x);
        }
      }
    }
  }
}

// ---------------- mean + 3-layer MLP head (tiny) ----------------

__global__ __launch_bounds__(256)
void k_mlp(const float* __restrict__ pool, const int* __restrict__ batch,
           const float* __restrict__ w1, const float* __restrict__ b1,
           const float* __restrict__ w2, const float* __restrict__ b2,
           const float* __restrict__ w3, const float* __restrict__ b3,
           float* __restrict__ out) {
  __shared__ int   s_cnt[BATCH];
  __shared__ float s_inv[BATCH];
  __shared__ float s_h1[BATCH][512];   // 32 KB
  __shared__ float s_h2[BATCH][256];   // 16 KB
  const int tid = threadIdx.x;
  if (tid < BATCH) s_cnt[tid] = 0;
  __syncthreads();
  for (int i = tid; i < N_PTS; i += 256) atomicAdd(&s_cnt[batch[i]], 1);
  __syncthreads();
  if (tid < BATCH) s_inv[tid] = 1.0f / fmaxf((float)s_cnt[tid], 1.0f);
  __syncthreads();

  for (int e = tid; e < BATCH * 512; e += 256) {
    int m = e >> 9, o = e & 511;
    float inv = s_inv[m];
    float s = b1[o];
    for (int i = 0; i < 1024; ++i) s += pool[m * 1024 + i] * inv * w1[i * 512 + o];
    s_h1[m][o] = fmaxf(s, 0.0f);
  }
  __syncthreads();
  for (int e = tid; e < BATCH * 256; e += 256) {
    int m = e >> 8, o = e & 255;
    float s = b2[o];
    for (int i = 0; i < 512; ++i) s += s_h1[m][i] * w2[i * 256 + o];
    s_h2[m][o] = fmaxf(s, 0.0f);
  }
  __syncthreads();
  for (int e = tid; e < BATCH * 152; e += 256) {
    int m = e / 152, o = e % 152;
    float s = b3[o];
    for (int i = 0; i < 256; ++i) s += s_h2[m][i] * w3[i * 152 + o];
    out[e] = s;
  }
}

// ---------------- launch ----------------

extern "C" void kernel_launch(void* const* d_in, const int* in_sizes, int n_in,
                              void* d_out, int out_size, void* d_ws, size_t ws_size,
                              hipStream_t stream) {
  (void)in_sizes; (void)n_in; (void)out_size; (void)ws_size;
  const float* pos   = (const float*)d_in[0];
  const float* feats = (const float*)d_in[1];
  const float* kp    = (const float*)d_in[2];
  const float* kpw   = (const float*)d_in[3];
  const float* w1    = (const float*)d_in[4];
  const float* b1    = (const float*)d_in[5];
  const float* w2    = (const float*)d_in[6];
  const float* b2    = (const float*)d_in[7];
  const float* w3    = (const float*)d_in[8];
  const float* b3    = (const float*)d_in[9];
  const int* nidx    = (const int*)d_in[10];
  const int* batch   = (const int*)d_in[11];

  char* ws = (char*)d_ws;
  unsigned short* featsb = (unsigned short*)(ws);                       // 6,400,000 B
  unsigned short* Wb     = (unsigned short*)(ws + 6400000);             // 2,097,152 B
  float*          pool   = (float*)(ws + 6400000 + 2097152);            //    65,536 B

  k_feats_bf16<<<(N_PTS * D_IN + 255) / 256, 256, 0, stream>>>(feats, featsb, N_PTS * D_IN);
  k_prep_w<<<(KK * D_OUT) / 256, 256, 0, stream>>>(kpw, Wb);
  k_zero<<<(BATCH * D_OUT + 255) / 256, 256, 0, stream>>>(pool, BATCH * D_OUT);
  k_kpconv<<<N_PTS / TILE, 128, 0, stream>>>(pos, kp, nidx, batch, featsb, Wb, pool);
  k_mlp<<<1, 256, 0, stream>>>(pool, batch, w1, b1, w2, b2, w3, b3, (float*)d_out);
}
